// QMoLE_Layer_68848325754901
// MI455X (gfx1250) — matrix operand order repaired and checked by reading, not compiled, not run on hardware
//
#include <hip/hip_runtime.h>

// ---------------------------------------------------------------------------
// MoE layer for MI455X (gfx1250).
// Memory-bound: x (64MB fp32 in) + out (64MB fp32 out) dominate; weights
// (~1.1MB bf16, L2-resident) stream from L2. Compute: v_wmma_f32_16x16x32_bf16
// with fp32 accumulate; router fused into the down-proj K-loop as a 9th slot.
// Two waves cooperate per 16-token tile (K-split down, h-split up) so the
// grid carries 1024 waves for latency hiding, with zero duplicated traffic.
// ---------------------------------------------------------------------------

#define HDIM 2048
#define NEXP 8
#define IDIM 16
#define KAPPA (NEXP * IDIM)   // 128 fused expert-intermediate dim

typedef __attribute__((ext_vector_type(16))) __bf16 v16bf;
typedef __attribute__((ext_vector_type(8)))  float  v8f;

union BFrag {
  v16bf v;
  uint4 q[2];
};
union V8U {
  v8f v;
  float4 f[2];
};

// workspace layout (bf16 elements):
//   wbuf [9 slots][16 cols][HDIM k]  : slots 0..7 = down_w[e][i][k] (identity
//                                      copy of down_w ordering), slot 8 =
//                                      router_w padded to 16 cols.
//   ubuf [HDIM h][KAPPA]             : ubuf[h][i*8+e] = up_w[e][h][i]
#define WBUF_ELEMS (9 * 16 * HDIM)          // 294912
#define UBUF_ELEMS (HDIM * KAPPA)           // 262144
#define PREP_TOTAL (WBUF_ELEMS + UBUF_ELEMS)

__global__ __launch_bounds__(256) void moe_prep(
    const float* __restrict__ router_w,
    const float* __restrict__ down_w,
    const float* __restrict__ up_w,
    __bf16* __restrict__ wsbuf) {
  int idx = blockIdx.x * 256 + threadIdx.x;
  if (idx < NEXP * IDIM * HDIM) {
    // down_w flat layout (e,i,k) == wbuf slot/col/k layout: direct convert
    wsbuf[idx] = (__bf16)down_w[idx];
  } else if (idx < WBUF_ELEMS) {
    int r = idx - NEXP * IDIM * HDIM;
    int col = r / HDIM;
    int k   = r % HDIM;
    wsbuf[idx] = (col < NEXP) ? (__bf16)router_w[col * HDIM + k] : (__bf16)0.0f;
  } else if (idx < PREP_TOTAL) {
    int r = idx - WBUF_ELEMS;
    int h   = r >> 7;       // / KAPPA
    int kap = r & 127;      // % KAPPA
    int i = kap >> 3;
    int e = kap & 7;
    wsbuf[idx] = (__bf16)up_w[((long)e * HDIM + h) * IDIM + i];
  }
}

// Two waves (64 threads) cooperate on one 16-token tile.
__global__ __launch_bounds__(64) void moe_main(
    const float* __restrict__ x,
    const __bf16* __restrict__ wbuf,
    const __bf16* __restrict__ ubuf,
    float* __restrict__ out) {
  __shared__ __align__(16) float part[9][16][16];    // wave-1 partials [s][n][t]
  __shared__ float lg[16][8];                        // router logits (transposed)
  __shared__ float gates[16][8];                     // top-2 masked softmax
  __shared__ __align__(16) __bf16 aprime[16][KAPPA]; // gate*silu(h), kappa=i*8+e

  const int tid  = threadIdx.x;
  const int wave = tid >> 5;
  const int lane = tid & 31;
  const int lo = lane & 15;
  const int hi = lane >> 4;
  const long t0 = (long)blockIdx.x * 16;

  v8f acc[9] = {};   // 8 expert down-proj tiles + router (slot 8)

  const float* xrow = x + (t0 + lo) * HDIM;

  // ---- fused down-projection + router, K split across the two waves ----
  const int kbeg = wave * (HDIM / 2);
  const int kend = kbeg + (HDIM / 2);
#pragma unroll 2
  for (int kc = kbeg; kc < kend; kc += 32) {
    // A fragment (16-bit A 16x32): lane(lo,hi) holds K = kc+hi*8+0..7 and
    // K = kc+16+hi*8+0..7 of token (t0+lo). fp32 loads, convert to bf16.
    float4 a0 = *(const float4*)(xrow + kc + hi * 8);
    float4 a1 = *(const float4*)(xrow + kc + hi * 8 + 4);
    float4 a2 = *(const float4*)(xrow + kc + 16 + hi * 8);
    float4 a3 = *(const float4*)(xrow + kc + 16 + hi * 8 + 4);
    BFrag A;
    v16bf av = {(__bf16)a0.x, (__bf16)a0.y, (__bf16)a0.z, (__bf16)a0.w,
                (__bf16)a1.x, (__bf16)a1.y, (__bf16)a1.z, (__bf16)a1.w,
                (__bf16)a2.x, (__bf16)a2.y, (__bf16)a2.z, (__bf16)a2.w,
                (__bf16)a3.x, (__bf16)a3.y, (__bf16)a3.z, (__bf16)a3.w};
    A.v = av;
#pragma unroll
    for (int s = 0; s < 9; s++) {
      // B fragment (16-bit B 32x16): lane = column lo, K = kc+hi*8+0..7
      // and K = kc+16+hi*8+0..7 (col-major weight buffer, two 16B loads).
      const __bf16* bp = wbuf + ((s * 16 + lo) * HDIM + kc + hi * 8);
      BFrag B;
      B.q[0] = *(const uint4*)bp;
      B.q[1] = *(const uint4*)(bp + 16);
      acc[s] = __builtin_amdgcn_wmma_f32_16x16x32_bf16(
          false, A.v, false, B.v, (short)0, acc[s], false, false);
    }
  }

  // ---- cross-wave reduction of the 9 partial C tiles via LDS ----
  // acc[s][r] holds element (t = r + 8*hi, n = lo); [s][n][t] layout makes
  // the 8 per-lane elements contiguous -> two b128 DS ops per slot.
  if (wave == 1) {
#pragma unroll
    for (int s = 0; s < 9; s++) {
      V8U u; u.v = acc[s];
      *(float4*)&part[s][lo][8 * hi]     = u.f[0];
      *(float4*)&part[s][lo][8 * hi + 4] = u.f[1];
    }
  }
  __syncthreads();
  if (wave == 0) {
#pragma unroll
    for (int s = 0; s < 9; s++) {
      float4 p0 = *(const float4*)&part[s][lo][8 * hi];
      float4 p1 = *(const float4*)&part[s][lo][8 * hi + 4];
      acc[s][0] += p0.x; acc[s][1] += p0.y;
      acc[s][2] += p0.z; acc[s][3] += p0.w;
      acc[s][4] += p1.x; acc[s][5] += p1.y;
      acc[s][6] += p1.z; acc[s][7] += p1.w;
    }
    // router logits -> LDS transpose
    if (lo < 8) {
#pragma unroll
      for (int r = 0; r < 8; r++) lg[r + 8 * hi][lo] = acc[8][r];
    }
    // softmax + top-2 gate per token (lanes 0..15, fp32)
    if (lane < 16) {
      float w[8];
      float m = lg[lane][0];
#pragma unroll
      for (int e = 1; e < 8; e++) m = fmaxf(m, lg[lane][e]);
      float ssum = 0.0f;
#pragma unroll
      for (int e = 0; e < 8; e++) { w[e] = __expf(lg[lane][e] - m); ssum += w[e]; }
      float inv = 1.0f / ssum;
#pragma unroll
      for (int e = 0; e < 8; e++) w[e] *= inv;
      int i1 = 0;                       // argmax, ties -> lowest index
#pragma unroll
      for (int e = 1; e < 8; e++) if (w[e] > w[i1]) i1 = e;
      int i2 = (i1 == 0) ? 1 : 0;       // 2nd argmax, ties -> lowest index
#pragma unroll
      for (int e = 0; e < 8; e++) if (e != i1 && w[e] > w[i2]) i2 = e;
#pragma unroll
      for (int e = 0; e < 8; e++)
        gates[lane][e] = (e == i1 || e == i2) ? w[e] : 0.0f;
    }
    // a' = gate * silu(h), packed bf16 to LDS (kappa = i*8 + e)
#pragma unroll
    for (int r = 0; r < 8; r++) {
      int t = r + 8 * hi;
      union { __bf16 b[8]; uint4 q; } pk;
#pragma unroll
      for (int e = 0; e < 8; e++) {
        float hv  = acc[e][r];                       // h[t, e, i=lo]
        float sil = hv / (1.0f + __expf(-hv));       // silu
        pk.b[e] = (__bf16)(gates[t][e] * sil);
      }
      *(uint4*)&aprime[t][lo * 8] = pk.q;            // 16B store, row t, i=lo
    }
  }
  __syncthreads();

  // ---- A2 fragments: a'[16 x 128] as four K=32 A-fragments (both waves) ----
  BFrag A2[4];
#pragma unroll
  for (int c = 0; c < 4; c++) {
    const __bf16* ap = &aprime[lo][c * 32 + hi * 8];
    A2[c].q[0] = *(const uint4*)ap;        // K = 32c + hi*8 + 0..7
    A2[c].q[1] = *(const uint4*)(ap + 16); // K = 32c + 16 + hi*8 + 0..7
  }

  // ---- up-projection, h-chunks split across the two waves ----
#pragma unroll 2
  for (int hc = wave * 16; hc < HDIM; hc += 32) {
    v8f d = {};
#pragma unroll
    for (int c = 0; c < 4; c++) {
      const __bf16* up = ubuf + ((hc + lo) * KAPPA + c * 32 + hi * 8);
      BFrag Bu;
      Bu.q[0] = *(const uint4*)up;
      Bu.q[1] = *(const uint4*)(up + 16);
      d = __builtin_amdgcn_wmma_f32_16x16x32_bf16(
          false, A2[c].v, false, Bu.v, (short)0, d, false, false);
    }
    // D layout: VGPR r -> (t = r + 8*hi, h = hc + lo); 16 consecutive floats
    // per lane-half per store coalesce in L2.
#pragma unroll
    for (int r = 0; r < 8; r++) {
      out[(t0 + r + 8 * hi) * HDIM + hc + lo] = d[r];
    }
  }
}

extern "C" void kernel_launch(void* const* d_in, const int* in_sizes, int n_in,
                              void* d_out, int out_size, void* d_ws, size_t ws_size,
                              hipStream_t stream) {
  const float* x        = (const float*)d_in[0];
  const float* router_w = (const float*)d_in[1];
  const float* down_w   = (const float*)d_in[2];
  const float* up_w     = (const float*)d_in[3];
  float* out = (float*)d_out;
  __bf16* wsbuf = (__bf16*)d_ws;   // needs PREP_TOTAL*2 = ~1.1MB scratch

  // 1) convert weights to WMMA-fragment-friendly bf16 layouts (L2-resident)
  moe_prep<<<(PREP_TOTAL + 255) / 256, 256, 0, stream>>>(router_w, down_w,
                                                         up_w, wsbuf);

  // 2) fused router + experts + combine; two waves per 16-token tile
  int ntok   = in_sizes[0] / HDIM;   // B*T = 8192
  int ntiles = ntok / 16;            // 512
  moe_main<<<ntiles, 64, 0, stream>>>(x, wsbuf, wsbuf + WBUF_ELEMS, out);
}
